// CrossCrissCrossAttention_78082505441649
// MI455X (gfx1250) — matrix operand order repaired and checked by reading, not compiled, not run on hardware
//
#include <hip/hip_runtime.h>
#include <cstddef>

#define BATCH 2
#define CH    64
#define HH    256
#define WW    256
#define HWSZ  (HH*WW)            // 65536
#define CQN   8
#define CRN   4
#define NS    8
#define NS2   16
#define NE    ((size_t)BATCH*CH*HWSZ)   // 8388608 elems
#define NQK   ((size_t)BATCH*16*HWSZ)   // 2097152 elems
#define NROW  ((size_t)BATCH*HH*WW)     // 131072 rows of 512 scores

typedef __attribute__((ext_vector_type(16))) __bf16 v16bf;
typedef __attribute__((ext_vector_type(8)))  float  v8f;

// ---- helpers -------------------------------------------------------------

// native f32 -> bf16 (RTNE in hardware on gfx1250)
__device__ inline __bf16 f2bf(float f) { return (__bf16)f; }

// 16-bit A-matrix 16x32: per-lane K index for vector slot j (0..15)
// (ISA 7.12.2: VGPR p<4 -> K=2p, p>=4 -> K=16+2(p-4); lanes 16-31 add 8)
__device__ inline int a_slot_k(int j, int lh) {
  int p = j >> 1;
  int base = (p < 4) ? (2 * p) : (16 + 2 * (p - 4));
  return base + (j & 1) + (lh ? 8 : 0);
}

__device__ inline float bred(float v, bool mx, float* sh) {
  #pragma unroll
  for (int off = 16; off > 0; off >>= 1)
    v = mx ? fmaxf(v, __shfl_xor(v, off, 32)) : (v + __shfl_xor(v, off, 32));
  int w = threadIdx.x >> 5;
  if ((threadIdx.x & 31) == 0) sh[w] = v;
  __syncthreads();
  float r = sh[0];
  for (int i = 1; i < (int)(blockDim.x >> 5); i++)
    r = mx ? fmaxf(r, sh[i]) : (r + sh[i]);
  __syncthreads();
  return r;
}

// ---- K0: pack weights (v|q|k -> 80x64 bf16 per side) ---------------------

__global__ void k_pack(const float* wv1, const float* wq1, const float* wk1,
                       const float* wv2, const float* wq2, const float* wk2,
                       const float* bv1, const float* bq1, const float* bk1,
                       const float* bv2, const float* bq2, const float* bk2,
                       __bf16* Wb, float* Bb) {
  int t = blockIdx.x * 256 + threadIdx.x;
  if (t < 2 * 80 * 64) {
    int side = t / (80 * 64); int o = (t / 64) % 80; int c = t % 64;
    const float* wv = side ? wv2 : wv1;
    const float* wq = side ? wq2 : wq1;
    const float* wk = side ? wk2 : wk1;
    float v;
    if (o < 64)      v = wv[o * CH + c];
    else if (o < 72) v = wq[(o - 64) * CH + c];
    else             v = wk[(o - 72) * CH + c];
    Wb[t] = f2bf(v);
  }
  if (t < 160) {
    int side = t / 80, o = t % 80;
    const float* bv = side ? bv2 : bv1;
    const float* bq = side ? bq2 : bq1;
    const float* bk = side ? bk2 : bk1;
    Bb[t] = (o < 64) ? bv[o] : (o < 72 ? bq[o - 64] : bk[o - 72]);
  }
}

// ---- K1: fused 1x1 conv, WMMA bf16: [80,64] x [64, HW] -------------------
// Outputs V (64 ch) and QK (8 q + 8 k ch) directly in bf16: they are only
// ever consumed as WMMA operands, so this halves their HBM traffic.

__global__ __launch_bounds__(256) void k_conv(const float* __restrict__ x,
    const __bf16* __restrict__ Wb, const float* __restrict__ Bb,
    __bf16* __restrict__ Vout, __bf16* __restrict__ QKout, int side) {
  int wave = threadIdx.x >> 5, lane = threadIdx.x & 31;
  int lh = lane >> 4, ln = lane & 15;
  int b = blockIdx.y;
  int n0 = blockIdx.x * 128 + wave * 16;
  const __bf16* Wp = Wb + side * 80 * 64;
  const float*  Bp = Bb + side * 80;

  v16bf bf[2];
  #pragma unroll
  for (int kb = 0; kb < 2; kb++) {
    #pragma unroll
    for (int j = 0; j < 16; j++) {
      int c = kb * 32 + lh * 16 + j;
      bf[kb][j] = f2bf(x[((size_t)b * CH + c) * HWSZ + n0 + ln]);
    }
  }
  for (int mt = 0; mt < 5; mt++) {
    v16bf a0, a1;
    #pragma unroll
    for (int j = 0; j < 16; j++) {
      int k = lh * 16 + j;
      a0[j] = Wp[(mt * 16 + ln) * 64 + k];
      a1[j] = Wp[(mt * 16 + ln) * 64 + 32 + k];
    }
    v8f acc;
    #pragma unroll
    for (int r = 0; r < 8; r++) acc[r] = Bp[mt * 16 + r + 8 * lh];
    acc = __builtin_amdgcn_wmma_f32_16x16x32_bf16(false, a0, false, bf[0], (short)0, acc, false, false);
    acc = __builtin_amdgcn_wmma_f32_16x16x32_bf16(false, a1, false, bf[1], (short)0, acc, false, false);
    #pragma unroll
    for (int r = 0; r < 8; r++) {
      int m = mt * 16 + r + 8 * lh;
      if (mt < 4) Vout[((size_t)b * CH + m) * HWSZ + n0 + ln] = f2bf(acc[r]);
      else        QKout[((size_t)b * 16 + (m - 64)) * HWSZ + n0 + ln] = f2bf(acc[r]);
    }
  }
}

// ---- K2: attention scores (K=8 zero-padded to 32) ------------------------
// mode 0: eH over column (fixed w), mode 1: eW over row (fixed h)

__global__ __launch_bounds__(256) void k_scores(const __bf16* __restrict__ Q,
    const __bf16* __restrict__ K, float* __restrict__ E, int mode) {
  int wave = threadIdx.x >> 5, lane = threadIdx.x & 31;
  int lh = lane >> 4, ln = lane & 15;
  int b = blockIdx.z, fixed = blockIdx.y;
  int mtile = blockIdx.x * 8 + wave;
  int r = mtile * 16 + ln;                   // h (mode0) or w (mode1)
  __bf16 z = f2bf(0.f);

  v16bf a;
  #pragma unroll
  for (int j = 0; j < 16; j++) {
    int c = a_slot_k(j, lh);
    __bf16 qv = z;
    if (c < CQN) {
      size_t idx = ((size_t)b * 16 + c) * HWSZ +
                   (mode == 0 ? (size_t)r * WW + fixed : (size_t)fixed * WW + r);
      qv = Q[idx];
    }
    a[j] = qv;
  }
  for (int nt = 0; nt < 16; nt++) {
    int col = nt * 16 + ln;                  // g (mode0) or v (mode1)
    v16bf bb;
    #pragma unroll
    for (int j = 0; j < 16; j++) {
      int c = a_slot_k(j, lh);
      __bf16 kv = z;
      if (c < CQN) {
        size_t idx = ((size_t)b * 16 + 8 + c) * HWSZ +
                     (mode == 0 ? (size_t)col * WW + fixed : (size_t)fixed * WW + col);
        kv = K[idx];
      }
      bb[j] = kv;
    }
    v8f acc;
    #pragma unroll
    for (int rr = 0; rr < 8; rr++) acc[rr] = 0.f;
    acc = __builtin_amdgcn_wmma_f32_16x16x32_bf16(false, a, false, bb, (short)0, acc, false, false);
    #pragma unroll
    for (int rr = 0; rr < 8; rr++) {
      int m = mtile * 16 + rr + 8 * lh;
      size_t eidx = (mode == 0)
        ? ((((size_t)b * HH + m) * WW + fixed) * 512 + col)
        : ((((size_t)b * HH + fixed) * WW + m) * 512 + 256 + col);
      E[eidx] = acc[rr];
    }
  }
}

// ---- K3: joint softmax over 512, diag mask on eH part; probs bf16 in-place

__global__ __launch_bounds__(256) void k_softmax(float* __restrict__ E) {
  int wave = threadIdx.x >> 5, lane = threadIdx.x & 31;
  size_t row = (size_t)blockIdx.x * 8 + wave;
  int h = (int)((row / WW) % HH);
  const float* er = E + row * 512;
  float v[16]; float mx = -3.4e38f;
  #pragma unroll
  for (int i = 0; i < 16; i++) {
    int idx = i * 32 + lane;
    float t = er[idx];
    if (idx == h) t = -3.4e38f;              // eye-mask (idx<256 when == h)
    v[i] = t; mx = fmaxf(mx, t);
  }
  #pragma unroll
  for (int off = 16; off > 0; off >>= 1) mx = fmaxf(mx, __shfl_xor(mx, off, 32));
  float s = 0.f;
  #pragma unroll
  for (int i = 0; i < 16; i++) { v[i] = __expf(v[i] - mx); s += v[i]; }
  #pragma unroll
  for (int off = 16; off > 0; off >>= 1) s += __shfl_xor(s, off, 32);
  float inv = 1.f / s;
  __bf16* pr = (__bf16*)E + row * 1024;      // first half of this row's bytes
  #pragma unroll
  for (int i = 0; i < 16; i++) pr[i * 32 + lane] = f2bf(v[i] * inv);
}

// ---- K4: aggregation out = A(probs) x V, WMMA bf16, K=256 ---------------
// mode 0: outH (fixed w, write), mode 1: outW (fixed h, accumulate)

__global__ __launch_bounds__(256) void k_agg(const __bf16* __restrict__ V,
    const float* __restrict__ E, float* __restrict__ ATT, int mode) {
  int wave = threadIdx.x >> 5, lane = threadIdx.x & 31;
  int lh = lane >> 4, ln = lane & 15;
  int b = blockIdx.z, fixed = blockIdx.y;
  int ntile = blockIdx.x * 8 + wave;
  int col = ntile * 16 + ln;                 // h (mode0) or w (mode1)

  v8f acc[4];
  #pragma unroll
  for (int mt = 0; mt < 4; mt++)
    #pragma unroll
    for (int r = 0; r < 8; r++) acc[mt][r] = 0.f;

  size_t prow = (mode == 0) ? (((size_t)b * HH + col) * WW + fixed)
                            : (((size_t)b * HH + fixed) * WW + col);
  const __bf16* prbase = (const __bf16*)E + prow * 1024 + (mode == 0 ? 0 : 256) + lh * 16;

  for (int kb = 0; kb < 8; kb++) {
    v16bf bb = *(const v16bf*)(prbase + kb * 32);   // 32B contiguous B-frag
    if (kb < 7) __builtin_prefetch(prbase + (kb + 1) * 32, 0, 1);
    #pragma unroll
    for (int mt = 0; mt < 4; mt++) {
      int c = mt * 16 + ln;
      v16bf a;
      #pragma unroll
      for (int j = 0; j < 16; j++) {
        int g = kb * 32 + lh * 16 + j;
        size_t idx = ((size_t)b * CH + c) * HWSZ +
                     (mode == 0 ? (size_t)g * WW + fixed : (size_t)fixed * WW + g);
        a[j] = V[idx];
      }
      acc[mt] = __builtin_amdgcn_wmma_f32_16x16x32_bf16(false, a, false, bb, (short)0, acc[mt], false, false);
    }
  }
  #pragma unroll
  for (int mt = 0; mt < 4; mt++) {
    #pragma unroll
    for (int r = 0; r < 8; r++) {
      int c = mt * 16 + r + 8 * lh;
      size_t idx = ((size_t)b * CH + c) * HWSZ +
                   (mode == 0 ? (size_t)col * WW + fixed : (size_t)fixed * WW + col);
      if (mode == 0) ATT[idx] = acc[mt][r];
      else           ATT[idx] += acc[mt][r];
    }
  }
}

// ---- P1: s = gamma*att + xa + xb; per-(b,c) sum/max partials -------------

__global__ __launch_bounds__(256) void k_spool(const float* __restrict__ x1,
    const float* __restrict__ x2, const float* __restrict__ att1,
    const float* __restrict__ att2, const float* g1p, const float* g2p,
    float* __restrict__ S1, float* __restrict__ S2, float* psum, float* pmax) {
  __shared__ float sh[8];
  int pc = blockIdx.x / NS;                  // b*CH + c
  int split = blockIdx.x % NS;
  float g1 = g1p[0], g2 = g2p[0];
  size_t base = (size_t)pc * HWSZ + (size_t)split * (HWSZ / NS) + threadIdx.x;
  float s1s = 0, s2s = 0, s1m = -3.4e38f, s2m = -3.4e38f;
  for (int i = 0; i < HWSZ / NS / 256; i++) {
    size_t idx = base + (size_t)i * 256;
    float a1 = x1[idx], a2 = x2[idx];
    float s1 = g2 * att2[idx] + a2 + a1;
    float s2 = g1 * att1[idx] + a1 + a2;
    S1[idx] = s1; S2[idx] = s2;
    s1s += s1; s2s += s2; s1m = fmaxf(s1m, s1); s2m = fmaxf(s2m, s2);
  }
  s1s = bred(s1s, false, sh); s2s = bred(s2s, false, sh);
  s1m = bred(s1m, true,  sh); s2m = bred(s2m, true,  sh);
  if (threadIdx.x == 0) {
    psum[pc * NS + split] = s1s;
    psum[BATCH * CH * NS + pc * NS + split] = s2s;
    pmax[pc * NS + split] = s1m;
    pmax[BATCH * CH * NS + pc * NS + split] = s2m;
  }
}

// ---- P2: finalize pools, tiny MLP, sigmoid gates -------------------------

__global__ void k_gate(const float* psum, const float* pmax,
    const float* w11, const float* b11, const float* w12, const float* b12,
    const float* w21, const float* b21, const float* w22, const float* b22,
    float* gate) {
  __shared__ float mn[2][BATCH][CH], mx[2][BATCH][CH];
  int t = threadIdx.x;                       // 256 threads
  int side = t >> 7, b = (t >> 6) & 1, c = t & 63;
  int pc = b * CH + c;
  const float* ps = psum + side * BATCH * CH * NS + pc * NS;
  const float* pm = pmax + side * BATCH * CH * NS + pc * NS;
  float s = 0.f, m = -3.4e38f;
  for (int i = 0; i < NS; i++) { s += ps[i]; m = fmaxf(m, pm[i]); }
  mn[side][b][c] = s / (float)HWSZ;
  mx[side][b][c] = m;
  __syncthreads();
  const float* w1 = side ? w21 : w11; const float* b1 = side ? b21 : b11;
  const float* w2 = side ? w22 : w12; const float* b2 = side ? b22 : b12;
  float v = 2.f * b2[c];                     // bias applied in both MLP branches
  #pragma unroll
  for (int r = 0; r < CRN; r++) {
    float hm = b1[r], hx = b1[r];
    for (int cc = 0; cc < CH; cc++) {
      hm += w1[r * CH + cc] * mn[side][b][cc];
      hx += w1[r * CH + cc] * mx[side][b][cc];
    }
    v += w2[c * CRN + r] * (fmaxf(hm, 0.f) + fmaxf(hx, 0.f));
  }
  gate[side * BATCH * CH + pc] = 1.f / (1.f + __expf(-v));
}

// ---- P3: t = gate*s + x; per-channel BN partial sums ---------------------

__global__ __launch_bounds__(256) void k_tbn(const float* __restrict__ x1,
    const float* __restrict__ x2, const float* __restrict__ S1,
    const float* __restrict__ S2, const float* __restrict__ gate,
    float* __restrict__ T1, float* __restrict__ T2, float* bsum, float* bsq) {
  __shared__ float sh[8];
  int c = blockIdx.x / NS2, split = blockIdx.x % NS2;
  size_t chunk = (size_t)BATCH * HWSZ / NS2;
  float s1 = 0, q1 = 0, s2 = 0, q2 = 0;
  for (int i = 0; i < (int)(chunk / 256); i++) {
    size_t j = (size_t)split * chunk + (size_t)i * 256 + threadIdx.x;
    int b = (int)(j / HWSZ); size_t k = j % HWSZ;
    size_t idx = ((size_t)b * CH + c) * HWSZ + k;
    float g1 = gate[b * CH + c];
    float g2 = gate[BATCH * CH + b * CH + c];
    float t1 = g1 * S1[idx] + x1[idx];
    float t2 = g2 * S2[idx] + x2[idx];
    T1[idx] = t1; T2[idx] = t2;
    s1 += t1; q1 += t1 * t1; s2 += t2; q2 += t2 * t2;
  }
  s1 = bred(s1, false, sh); q1 = bred(q1, false, sh);
  s2 = bred(s2, false, sh); q2 = bred(q2, false, sh);
  if (threadIdx.x == 0) {
    bsum[c * NS2 + split] = s1;             bsq[c * NS2 + split] = q1;
    bsum[CH * NS2 + c * NS2 + split] = s2;  bsq[CH * NS2 + c * NS2 + split] = q2;
  }
}

// ---- P4: BN coefficients -------------------------------------------------

__global__ void k_bncoef(const float* bsum, const float* bsq,
    const float* sc1, const float* bi1, const float* sc2, const float* bi2,
    float* alpha, float* beta) {
  int t = threadIdx.x; if (t >= 2 * CH) return;
  int side = t >> 6, c = t & 63;
  const float* su = bsum + side * CH * NS2 + c * NS2;
  const float* sq = bsq  + side * CH * NS2 + c * NS2;
  float s = 0, q = 0;
  for (int i = 0; i < NS2; i++) { s += su[i]; q += sq[i]; }
  float n = (float)((size_t)BATCH * HWSZ);
  float mean = s / n, var = q / n - mean * mean;
  float sc = side ? sc2[c] : sc1[c];
  float bi = side ? bi2[c] : bi1[c];
  float a = sc * rsqrtf(var + 1e-5f);
  alpha[side * CH + c] = a;
  beta[side * CH + c] = bi - mean * a;
}

// ---- P5: BN-ReLU + even-channel passthrough mask -------------------------

__global__ __launch_bounds__(256) void k_out(const float* __restrict__ x1,
    const float* __restrict__ x2, const float* __restrict__ T1,
    const float* __restrict__ T2, const float* __restrict__ alpha,
    const float* __restrict__ beta, float* __restrict__ out) {
  size_t i0 = ((size_t)blockIdx.x * 256 + threadIdx.x) * 4;
  #pragma unroll
  for (int u = 0; u < 4; u++) {
    size_t idx = i0 + u; if (idx >= NE) return;
    int c = (int)((idx / HWSZ) % CH);
    bool keep = (c & 1) == 0;
    float o1 = keep ? x1[idx] : fmaxf(T1[idx] * alpha[c] + beta[c], 0.f);
    float o2 = keep ? x2[idx] : fmaxf(T2[idx] * alpha[CH + c] + beta[CH + c], 0.f);
    out[idx] = o1;
    out[NE + idx] = o2;
  }
}

// ---- launcher ------------------------------------------------------------

extern "C" void kernel_launch(void* const* d_in, const int* in_sizes, int n_in,
                              void* d_out, int out_size, void* d_ws, size_t ws_size,
                              hipStream_t stream) {
  const float* x1  = (const float*)d_in[0];
  const float* x2  = (const float*)d_in[1];
  const float* wq1 = (const float*)d_in[2];  const float* bq1 = (const float*)d_in[3];
  const float* wq2 = (const float*)d_in[4];  const float* bq2 = (const float*)d_in[5];
  const float* wk1 = (const float*)d_in[6];  const float* bk1 = (const float*)d_in[7];
  const float* wk2 = (const float*)d_in[8];  const float* bk2 = (const float*)d_in[9];
  const float* wv1 = (const float*)d_in[10]; const float* bv1 = (const float*)d_in[11];
  const float* wv2 = (const float*)d_in[12]; const float* bv2 = (const float*)d_in[13];
  const float* g1  = (const float*)d_in[14]; const float* g2  = (const float*)d_in[15];
  const float* s1w1 = (const float*)d_in[16]; const float* s1b1 = (const float*)d_in[17];
  const float* s1w2 = (const float*)d_in[18]; const float* s1b2 = (const float*)d_in[19];
  const float* s2w1 = (const float*)d_in[20]; const float* s2b1 = (const float*)d_in[21];
  const float* s2w2 = (const float*)d_in[22]; const float* s2b2 = (const float*)d_in[23];
  const float* bn1s = (const float*)d_in[24]; const float* bn1b = (const float*)d_in[25];
  const float* bn2s = (const float*)d_in[26]; const float* bn2b = (const float*)d_in[27];

  char* ws = (char*)d_ws;
  size_t off = 0;
  auto take = [&](size_t bytes) -> char* {
    char* p = ws + off; off = (off + bytes + 255) & ~(size_t)255; return p;
  };
  __bf16* V1  = (__bf16*)take(NE * 2);              // bf16 V
  __bf16* V2  = (__bf16*)take(NE * 2);
  __bf16* QK1 = (__bf16*)take(NQK * 2);             // bf16 q|k
  __bf16* QK2 = (__bf16*)take(NQK * 2);
  float* E    = (float*)take(NROW * 512 * 4);       // 268 MB, reused across passes
  float* ATT1 = (float*)take(NE * 4);
  float* ATT2 = (float*)take(NE * 4);
  // aliases: S reuses E (E fully consumed by then), T reuses ATT
  float* S1 = E;  float* S2 = E + NE;
  float* T1 = ATT1; float* T2 = ATT2;
  __bf16* Wb  = (__bf16*)take(2 * 80 * 64 * 2);
  float* Bb   = (float*)take(2 * 80 * 4);
  float* GATE = (float*)take(2 * BATCH * CH * 4);
  float* PSUM = (float*)take(2 * BATCH * CH * NS * 4);
  float* PMAX = (float*)take(2 * BATCH * CH * NS * 4);
  float* BSUM = (float*)take(2 * CH * NS2 * 4);
  float* BSQ  = (float*)take(2 * CH * NS2 * 4);
  float* ALPH = (float*)take(2 * CH * 4);
  float* BETA = (float*)take(2 * CH * 4);

  k_pack<<<40, 256, 0, stream>>>(wv1, wq1, wk1, wv2, wq2, wk2,
                                 bv1, bq1, bk1, bv2, bq2, bk2, Wb, Bb);
  k_conv<<<dim3(HWSZ / 128, BATCH), 256, 0, stream>>>(x1, Wb, Bb, V1, QK1, 0);
  k_conv<<<dim3(HWSZ / 128, BATCH), 256, 0, stream>>>(x2, Wb, Bb, V2, QK2, 1);

  // att1 = criss_cross(q2, k1, v1)
  k_scores<<<dim3(2, WW, BATCH), 256, 0, stream>>>(QK2, QK1, E, 0);
  k_scores<<<dim3(2, HH, BATCH), 256, 0, stream>>>(QK2, QK1, E, 1);
  k_softmax<<<(unsigned)(NROW / 8), 256, 0, stream>>>(E);
  k_agg<<<dim3(2, WW, BATCH), 256, 0, stream>>>(V1, E, ATT1, 0);
  k_agg<<<dim3(2, HH, BATCH), 256, 0, stream>>>(V1, E, ATT1, 1);

  // att2 = criss_cross(q1, k2, v2) -- reuse E
  k_scores<<<dim3(2, WW, BATCH), 256, 0, stream>>>(QK1, QK2, E, 0);
  k_scores<<<dim3(2, HH, BATCH), 256, 0, stream>>>(QK1, QK2, E, 1);
  k_softmax<<<(unsigned)(NROW / 8), 256, 0, stream>>>(E);
  k_agg<<<dim3(2, WW, BATCH), 256, 0, stream>>>(V2, E, ATT2, 0);
  k_agg<<<dim3(2, HH, BATCH), 256, 0, stream>>>(V2, E, ATT2, 1);

  k_spool<<<BATCH * CH * NS, 256, 0, stream>>>(x1, x2, ATT1, ATT2, g1, g2,
                                               S1, S2, PSUM, PMAX);
  k_gate<<<1, 256, 0, stream>>>(PSUM, PMAX, s1w1, s1b1, s1w2, s1b2,
                                s2w1, s2b1, s2w2, s2b2, GATE);
  k_tbn<<<CH * NS2, 256, 0, stream>>>(x1, x2, S1, S2, GATE, T1, T2, BSUM, BSQ);
  k_bncoef<<<1, 128, 0, stream>>>(BSUM, BSQ, bn1s, bn1b, bn2s, bn2b, ALPH, BETA);
  k_out<<<(unsigned)(NE / 1024), 256, 0, stream>>>(x1, x2, T1, T2, ALPH, BETA,
                                                   (float*)d_out);
}
